// SparseMoEOptimized_54795192763069
// MI455X (gfx1250) — matrix operand order repaired and checked by reading, not compiled, not run on hardware
//
#include <hip/hip_runtime.h>
#include <hip/hip_bf16.h>
#include <stddef.h>

// Problem constants (match reference)
#define N_EMBD 1024
#define D_FF   4096
#define NUM_E  8
#define TOKENS 8192          // B*S = 4*2048
#define MT     32            // tokens per expert tile (two 16-row WMMA tiles)
#define TPB    256           // 8 wave32s per workgroup

// Pre-swizzled bf16 weight geometry: [E][nsub][kstep][lane][16 bf16]
#define W1_NSUB (D_FF / 16)    // 256
#define W1_KST  (N_EMBD / 32)  // 32
#define W2_NSUB (N_EMBD / 16)  // 64
#define W2_KST  (D_FF / 32)    // 128
#define W1_ENT  (NUM_E * W1_NSUB * W1_KST * 32)  // v16bf entries
#define W2_ENT  (NUM_E * W2_NSUB * W2_KST * 32)

typedef __attribute__((ext_vector_type(16))) __bf16 v16bf;
typedef __attribute__((ext_vector_type(8)))  float  v8f;

#define WMMA_BF16(a, b, c) \
  __builtin_amdgcn_wmma_f32_16x16x32_bf16(false, (a), false, (b), (short)0, (c), false, false)

// ---------------------------------------------------------------------------
// A-fragment (16-bit, 16x32) LDS index for element (m, k) of a [32 x K] tile
// stored as [mtile][kstep][lane][16 bf16] (contiguous k = 16*half + i within each kstep)
// ---------------------------------------------------------------------------
__device__ __forceinline__ int hfrag_idx(int m, int k, int ksteps) {
  int mt  = m >> 4, mm = m & 15;
  int kk  = k >> 5, kin = k & 31;
  int half, i;
  half = kin >> 4; i = kin & 15;
  return ((mt * ksteps + kk) * 32 + (16 * half + mm)) * 16 + i;
}

// Phase-1 epilogue: bias + relu, store 16 C-fragment rows into LDS h (A layout)
__device__ __forceinline__ void store_h_tile(__bf16* hs, const v8f& acc0, const v8f& acc1,
                                             float bias, int n, int half) {
  #pragma unroll
  for (int v = 0; v < 8; ++v) {
    float h0 = fmaxf(acc0[v] + bias, 0.0f);
    float h1 = fmaxf(acc1[v] + bias, 0.0f);
    int m0 = v + 8 * half;
    hs[hfrag_idx(m0,      n, 128)] = (__bf16)h0;
    hs[hfrag_idx(m0 + 16, n, 128)] = (__bf16)h1;
  }
}

// Phase-2 epilogue: bias, router-weight scale, fp32 atomic scatter into out
__device__ __forceinline__ void scatter_out(float* out, const int* lst, const float* rw,
                                            const v8f& acc0, const v8f& acc1, float bias,
                                            int n, int half, int base, int count, int e) {
  #pragma unroll
  for (int v = 0; v < 8; ++v) {
    int m0 = v + 8 * half;
    int r0 = base + m0;
    if (r0 < count) {
      int tok   = lst[r0];
      float wgt = rw[tok * NUM_E + e];
      __hip_atomic_fetch_add(out + (size_t)tok * N_EMBD + n, (acc0[v] + bias) * wgt,
                             __ATOMIC_RELAXED, __HIP_MEMORY_SCOPE_AGENT);
    }
    int r1 = base + m0 + 16;
    if (r1 < count) {
      int tok   = lst[r1];
      float wgt = rw[tok * NUM_E + e];
      __hip_atomic_fetch_add(out + (size_t)tok * N_EMBD + n, (acc1[v] + bias) * wgt,
                             __ATOMIC_RELAXED, __HIP_MEMORY_SCOPE_AGENT);
    }
  }
}

// ---------------------------------------------------------------------------
// Kernel 1: zero output accumulator and per-expert counters
// ---------------------------------------------------------------------------
__global__ __launch_bounds__(TPB) void moe_zero_kernel(float* __restrict__ out,
                                                       int n, int* __restrict__ counts) {
  int i = blockIdx.x * blockDim.x + threadIdx.x;
  if (i < NUM_E) counts[i] = 0;
  for (int idx = i; idx < n; idx += gridDim.x * blockDim.x) out[idx] = 0.0f;
}

// ---------------------------------------------------------------------------
// Kernel 2: router — logits, top-2 (ties -> lowest index), sparse softmax,
// compaction into per-expert token lists.
// ---------------------------------------------------------------------------
__global__ __launch_bounds__(TPB) void moe_router_kernel(
    const float* __restrict__ x, const float* __restrict__ Wr,
    const float* __restrict__ br, float* __restrict__ rw,
    int* __restrict__ counts, int* __restrict__ list) {
  int t = blockIdx.x * blockDim.x + threadIdx.x;
  if (t >= TOKENS) return;

  float acc[NUM_E];
#pragma unroll
  for (int e = 0; e < NUM_E; ++e) acc[e] = br[e];

  const float* xp = x + (size_t)t * N_EMBD;
  for (int k = 0; k < N_EMBD; ++k) {
    float xv = xp[k];
    const float4* w4 = (const float4*)(Wr + k * NUM_E);
    float4 wa = w4[0], wb = w4[1];
    acc[0] += xv * wa.x; acc[1] += xv * wa.y; acc[2] += xv * wa.z; acc[3] += xv * wa.w;
    acc[4] += xv * wb.x; acc[5] += xv * wb.y; acc[6] += xv * wb.z; acc[7] += xv * wb.w;
  }

  int i1 = 0;
#pragma unroll
  for (int e = 1; e < NUM_E; ++e) if (acc[e] > acc[i1]) i1 = e;
  int i2 = (i1 == 0) ? 1 : 0;
#pragma unroll
  for (int e = 0; e < NUM_E; ++e) if (e != i1 && acc[e] > acc[i2]) i2 = e;

  float v1 = acc[i1], v2 = acc[i2];
  float e2 = __expf(v2 - v1);
  float inv = 1.0f / (1.0f + e2);

#pragma unroll
  for (int e = 0; e < NUM_E; ++e) rw[t * NUM_E + e] = 0.0f;
  rw[t * NUM_E + i1] = inv;
  rw[t * NUM_E + i2] = e2 * inv;

  int p1 = atomicAdd(counts + i1, 1); list[i1 * TOKENS + p1] = t;
  int p2 = atomicAdd(counts + i2, 1); list[i2 * TOKENS + p2] = t;
}

// ---------------------------------------------------------------------------
// Kernel 2b: convert W1/W2 fp32 -> bf16, pre-swizzled into B-fragment order:
//   entry(e, nsub, kk, lane) = 16 bf16 with k = kk*32 + 16*(lane>=16) + i,
//   n = nsub*16 + lane%16.  One 32B coalesced store per thread.
// ---------------------------------------------------------------------------
__global__ __launch_bounds__(TPB) void moe_convert_kernel(
    const float* __restrict__ W1, const float* __restrict__ W2,
    v16bf* __restrict__ wb1, v16bf* __restrict__ wb2) {
  int ent = blockIdx.x * blockDim.x + threadIdx.x;
  if (ent < W1_ENT) {
    int lane = ent & 31;
    int kk   = (ent >> 5) % W1_KST;
    int nsub = (ent >> 5) / W1_KST % W1_NSUB;
    int e    = ent / (32 * W1_KST * W1_NSUB);
    int n    = nsub * 16 + (lane & 15);
    int kb   = kk * 32 + 16 * (lane >> 4);
    const float* src = W1 + (size_t)e * N_EMBD * D_FF + (size_t)kb * D_FF + n;
    v16bf v;
    #pragma unroll
    for (int i = 0; i < 16; ++i) v[i] = (__bf16)src[(size_t)i * D_FF];
    wb1[ent] = v;
  } else {
    int ent2 = ent - W1_ENT;
    if (ent2 >= W2_ENT) return;
    int lane = ent2 & 31;
    int kk   = (ent2 >> 5) % W2_KST;
    int nsub = (ent2 >> 5) / W2_KST % W2_NSUB;
    int e    = ent2 / (32 * W2_KST * W2_NSUB);
    int n    = nsub * 16 + (lane & 15);
    int kb   = kk * 32 + 16 * (lane >> 4);
    const float* src = W2 + (size_t)e * D_FF * N_EMBD + (size_t)kb * N_EMBD + n;
    v16bf v;
    #pragma unroll
    for (int i = 0; i < 16; ++i) v[i] = (__bf16)src[(size_t)i * N_EMBD];
    wb2[ent2] = v;
  }
}

// ---------------------------------------------------------------------------
// Kernel 3: fused expert MLP on a 32-token tile of one expert.
//   LDS: xs = x tile  [2][32][32] v16bf              ( 64 KB)
//        hs = h tile  [2][128][32][16] bf16          (256 KB)
//   Phase 1: h = relu(x @ W1[e] + b1[e])   (bf16 WMMA, fp32 acc)
//   Phase 2: out += (h @ W2[e] + b2[e]) * router_weight (fp32 atomics)
//   Fast path (wb1 != null): pre-swizzled bf16 B fragments, 2 n-subtiles per
//   k-loop so each A pair feeds 4 WMMAs (4 ds + 4 vmem + 4 wmma per step).
// ---------------------------------------------------------------------------
__global__ __launch_bounds__(TPB) void moe_expert_tile_kernel(
    const float* __restrict__ x,  const float* __restrict__ W1,
    const float* __restrict__ b1, const float* __restrict__ W2,
    const float* __restrict__ b2, const float* __restrict__ rw,
    const int* __restrict__ counts, const int* __restrict__ list,
    const v16bf* __restrict__ wb1, const v16bf* __restrict__ wb2,
    float* __restrict__ out) {
  extern __shared__ __align__(32) unsigned char smem[];
  v16bf*       xs = (v16bf*)smem;                                   // 2*32*32 v16bf
  __bf16*      hs = (__bf16*)(smem + 2 * 32 * 32 * sizeof(v16bf));  // 2*128*32*16 bf16
  const v16bf* hv = (const v16bf*)hs;

  const int e     = blockIdx.y;
  const int count = counts[e];
  const int base  = blockIdx.x * MT;
  if (base >= count) return;
  const int* lst = list + e * TOKENS;

  const int tid  = threadIdx.x;
  const int lane = tid & 31;
  const int wv   = tid >> 5;     // wave id 0..7
  const int half = lane >> 4;
  const int nlan = lane & 15;

  // ---- Phase 0: stage x tile into LDS, pre-swizzled into A-fragment layout
  #pragma unroll
  for (int j = 0; j < 8; ++j) {
    int ent = tid + j * TPB;          // 0..2047 = [2 mtiles][32 ksteps][32 lanes]
    int mt  = ent >> 10;
    int rem = ent & 1023;
    int kk  = rem >> 5;
    int ln  = rem & 31;
    int hf  = ln >> 4;
    int m   = mt * 16 + (ln & 15);
    int row = base + m; if (row >= count) row = count - 1;
    int tok = lst[row];
    const float* xp = x + (size_t)tok * N_EMBD + kk * 32 + 16 * hf;
    float4 fa = *(const float4*)(xp);
    float4 fb = *(const float4*)(xp + 4);
    float4 fc = *(const float4*)(xp + 8);
    float4 fd = *(const float4*)(xp + 12);
    v16bf v;
    v[0]  = (__bf16)fa.x; v[1]  = (__bf16)fa.y; v[2]  = (__bf16)fa.z; v[3]  = (__bf16)fa.w;
    v[4]  = (__bf16)fb.x; v[5]  = (__bf16)fb.y; v[6]  = (__bf16)fb.z; v[7]  = (__bf16)fb.w;
    v[8]  = (__bf16)fc.x; v[9]  = (__bf16)fc.y; v[10] = (__bf16)fc.z; v[11] = (__bf16)fc.w;
    v[12] = (__bf16)fd.x; v[13] = (__bf16)fd.y; v[14] = (__bf16)fd.z; v[15] = (__bf16)fd.w;
    xs[(mt * 32 + kk) * 32 + ln] = v;
  }
  __syncthreads();

  // ---- Phase 1: h = relu(x @ W1 + b1); each wave owns 512 of the 4096 cols
  const float* W1e = W1 + (size_t)e * N_EMBD * D_FF;
  if (wb1) {
    #pragma unroll 1
    for (int ns = 0; ns < 32; ns += 2) {
      const int nA = wv * 512 + ns * 16 + nlan;
      v8f acc0A = {}, acc1A = {}, acc0B = {}, acc1B = {};
      const v16bf* wbA = wb1 + ((size_t)(e * W1_NSUB + wv * 32 + ns) * W1_KST) * 32 + lane;
      const v16bf* wbB = wbA + (size_t)W1_KST * 32;
      #pragma unroll 4
      for (int kk = 0; kk < 32; ++kk) {
        v16bf a0 = xs[kk * 32 + lane];
        v16bf a1 = xs[(32 + kk) * 32 + lane];
        v16bf bA = wbA[kk * 32];
        v16bf bB = wbB[kk * 32];
        if ((kk & 3) == 0) {
          __builtin_prefetch(wbA + (kk + 16) * 32, 0, 1);
          __builtin_prefetch(wbB + (kk + 16) * 32, 0, 1);
        }
        acc0A = WMMA_BF16(a0, bA, acc0A);
        acc1A = WMMA_BF16(a1, bA, acc1A);
        acc0B = WMMA_BF16(a0, bB, acc0B);
        acc1B = WMMA_BF16(a1, bB, acc1B);
      }
      store_h_tile(hs, acc0A, acc1A, b1[e * D_FF + nA],      nA,      half);
      store_h_tile(hs, acc0B, acc1B, b1[e * D_FF + nA + 16], nA + 16, half);
    }
  } else {
    #pragma unroll 1
    for (int ns = 0; ns < 32; ++ns) {
      const int n = wv * 512 + ns * 16 + nlan;
      v8f acc0 = {}, acc1 = {};
      #pragma unroll 4
      for (int kk = 0; kk < 32; ++kk) {
        v16bf a0 = xs[kk * 32 + lane];
        v16bf a1 = xs[(32 + kk) * 32 + lane];
        const float* bp = W1e + (size_t)(kk * 32 + 16 * half) * D_FF + n;
        __builtin_prefetch(bp + 32 * D_FF, 0, 1);
        v16bf b;
        #pragma unroll
        for (int i = 0; i < 16; ++i) b[i] = (__bf16)bp[(size_t)i * D_FF];
        acc0 = WMMA_BF16(a0, b, acc0);
        acc1 = WMMA_BF16(a1, b, acc1);
      }
      store_h_tile(hs, acc0, acc1, b1[e * D_FF + n], n, half);
    }
  }
  __syncthreads();

  // ---- Phase 2: out += (h @ W2 + b2) * w; each wave owns 128 of 1024 cols
  const float* W2e = W2 + (size_t)e * D_FF * N_EMBD;
  if (wb2) {
    #pragma unroll 1
    for (int ns = 0; ns < 8; ns += 2) {
      const int nA = wv * 128 + ns * 16 + nlan;
      v8f acc0A = {}, acc1A = {}, acc0B = {}, acc1B = {};
      const v16bf* wbA = wb2 + ((size_t)(e * W2_NSUB + wv * 8 + ns) * W2_KST) * 32 + lane;
      const v16bf* wbB = wbA + (size_t)W2_KST * 32;
      #pragma unroll 4
      for (int kk = 0; kk < 128; ++kk) {
        v16bf a0 = hv[kk * 32 + lane];
        v16bf a1 = hv[(128 + kk) * 32 + lane];
        v16bf bA = wbA[kk * 32];
        v16bf bB = wbB[kk * 32];
        if ((kk & 3) == 0) {
          __builtin_prefetch(wbA + (kk + 16) * 32, 0, 1);
          __builtin_prefetch(wbB + (kk + 16) * 32, 0, 1);
        }
        acc0A = WMMA_BF16(a0, bA, acc0A);
        acc1A = WMMA_BF16(a1, bA, acc1A);
        acc0B = WMMA_BF16(a0, bB, acc0B);
        acc1B = WMMA_BF16(a1, bB, acc1B);
      }
      scatter_out(out, lst, rw, acc0A, acc1A, b2[e * N_EMBD + nA],      nA,      half, base, count, e);
      scatter_out(out, lst, rw, acc0B, acc1B, b2[e * N_EMBD + nA + 16], nA + 16, half, base, count, e);
    }
  } else {
    #pragma unroll 1
    for (int ns = 0; ns < 8; ++ns) {
      const int n = wv * 128 + ns * 16 + nlan;
      v8f acc0 = {}, acc1 = {};
      #pragma unroll 2
      for (int kk = 0; kk < 128; ++kk) {
        v16bf a0 = hv[kk * 32 + lane];
        v16bf a1 = hv[(128 + kk) * 32 + lane];
        const float* bp = W2e + (size_t)(kk * 32 + 16 * half) * N_EMBD + n;
        __builtin_prefetch(bp + 32 * N_EMBD, 0, 1);
        v16bf b;
        #pragma unroll
        for (int i = 0; i < 16; ++i) b[i] = (__bf16)bp[(size_t)i * N_EMBD];
        acc0 = WMMA_BF16(a0, b, acc0);
        acc1 = WMMA_BF16(a1, b, acc1);
      }
      scatter_out(out, lst, rw, acc0, acc1, b2[e * N_EMBD + n], n, half, base, count, e);
    }
  }
}

// ---------------------------------------------------------------------------
// Host-side launcher (graph-capture safe: launches only, all on `stream`)
// ---------------------------------------------------------------------------
extern "C" void kernel_launch(void* const* d_in, const int* in_sizes, int n_in,
                              void* d_out, int out_size, void* d_ws, size_t ws_size,
                              hipStream_t stream) {
  (void)in_sizes; (void)n_in; (void)out_size;
  const float* x  = (const float*)d_in[0];
  const float* Wr = (const float*)d_in[1];
  const float* br = (const float*)d_in[2];
  const float* W1 = (const float*)d_in[3];
  const float* b1 = (const float*)d_in[4];
  const float* W2 = (const float*)d_in[5];
  const float* b2 = (const float*)d_in[6];
  float* out = (float*)d_out;

  // Workspace: [rw: T*8 f32][counts: 8 i32 + pad][list: 8*T i32] then
  //            1MB-aligned [wb1: 64MB bf16][wb2: 64MB bf16] (optional)
  unsigned char* ws = (unsigned char*)d_ws;
  float* rw     = (float*)ws;
  int*   counts = (int*)(ws + (size_t)TOKENS * NUM_E * sizeof(float));
  int*   list   = (int*)(ws + (size_t)TOKENS * NUM_E * sizeof(float) + 128);
  size_t wboff  = (size_t)1 << 20;
  v16bf* wb1    = (v16bf*)(ws + wboff);
  v16bf* wb2    = wb1 + W1_ENT;
  size_t need   = wboff + ((size_t)W1_ENT + W2_ENT) * sizeof(v16bf);
  bool   useBf  = (ws_size >= need);

  moe_zero_kernel<<<dim3(4096), dim3(TPB), 0, stream>>>(out, TOKENS * N_EMBD, counts);
  moe_router_kernel<<<dim3(TOKENS / TPB), dim3(TPB), 0, stream>>>(x, Wr, br, rw, counts, list);
  if (useBf) {
    int nconv = (W1_ENT + W2_ENT + TPB - 1) / TPB;
    moe_convert_kernel<<<dim3(nconv), dim3(TPB), 0, stream>>>(W1, W2, wb1, wb2);
  }

  size_t shmem = (size_t)(2 * 32 * 32 * sizeof(v16bf))         // xs: 64 KB
               + (size_t)(2 * 128 * 32 * 16 * sizeof(__bf16)); // hs: 256 KB
  moe_expert_tile_kernel<<<dim3(TOKENS / MT, NUM_E), dim3(TPB), shmem, stream>>>(
      x, W1, b1, W2, b2, rw, counts, list,
      useBf ? wb1 : (const v16bf*)nullptr,
      useBf ? wb2 : (const v16bf*)nullptr, out);
}
